// DipNetEncoder_72189810311186
// MI455X (gfx1250) — matrix operand order repaired and checked by reading, not compiled, not run on hardware
//
#include <hip/hip_runtime.h>
#include <hip/hip_bf16.h>

// ---------------------------------------------------------------------------
// DipNet encoder for MI455X (gfx1250, wave32, WMMA).
// Per-node GEMMs + adjacency mix run as v_wmma_f32_16x16x32_bf16 (f32 accum).
// BN stats fused into the mix kernel via wave shfl reduction + f32 atomics.
// ---------------------------------------------------------------------------

typedef __bf16 bf16_t;
typedef __attribute__((ext_vector_type(8)))  __bf16 v8bf;
typedef __attribute__((ext_vector_type(16))) __bf16 v16bf;
typedef __attribute__((ext_vector_type(8)))  float  v8f;

#define BATCH 2048
#define NNODE 81
#define NPAD  96
#define EMB   256
#define NBLK  8

static __device__ __forceinline__ v16bf load_frag16(const bf16_t* p0, const bf16_t* p1) {
    v8bf lo = *reinterpret_cast<const v8bf*>(p0);   // 16B load
    v8bf hi = *reinterpret_cast<const v8bf*>(p1);   // 16B load
    return __builtin_shufflevector(lo, hi, 0,1,2,3,4,5,6,7,8,9,10,11,12,13,14,15);
}

// ---------------------------------------------------------------------------
// Conversion kernels (run once per launch; keep the hot loop memory-lean).
// ---------------------------------------------------------------------------

// A (81x81) f32 -> bf16 zero-padded 96x96 (zeros kill K/M padding in mix GEMM)
__global__ void convert_adj_kernel(const float* __restrict__ A, bf16_t* __restrict__ adjb) {
    int t = blockIdx.x * blockDim.x + threadIdx.x;
    if (t < NPAD * NPAD) {
        int nn = t % NPAD, mm = t / NPAD;
        float v = (mm < NNODE && nn < NNODE) ? A[mm * NNODE + nn] : 0.0f;
        adjb[t] = (bf16_t)v;
    }
}

// W0 (81, IN, 256) f32 -> bf16 transposed/padded: w0t[n][o][i], i padded to 64
__global__ void convert_w0_kernel(const float* __restrict__ w0, bf16_t* __restrict__ w0t, int in_dim) {
    int t = blockIdx.x * blockDim.x + threadIdx.x;    // 81*256 threads
    int o = t & 255, n = t >> 8;
    const float* src = w0 + (size_t)n * in_dim * EMB + o;
    bf16_t* dst = w0t + (size_t)t * 64;
    for (int i = 0; i < 64; ++i)
        dst[i] = (bf16_t)((i < in_dim) ? src[(size_t)i * EMB] : 0.0f);
}

// W (7,81,256,256) f32 [i][o] -> bf16 wt[(k*81+n)][o][i] via LDS-tiled transpose
__global__ __launch_bounds__(256) void convert_w_kernel(const float* __restrict__ w, bf16_t* __restrict__ wt) {
    __shared__ bf16_t lds[16 * EMB];
    int kn = blockIdx.y;                // (k*81+n)
    int it = blockIdx.x;                // i-tile (16 rows)
    const float* src = w + ((size_t)kn * EMB + it * 16) * EMB;
    #pragma unroll
    for (int r = 0; r < 16; ++r)
        lds[r * EMB + threadIdx.x] = (bf16_t)src[(size_t)r * EMB + threadIdx.x];
    __syncthreads();
    int o = threadIdx.x;
    bf16_t tmp[16];
    #pragma unroll
    for (int r = 0; r < 16; ++r) tmp[r] = lds[r * EMB + o];
    bf16_t* dst = wt + ((size_t)kn * EMB + o) * EMB + it * 16;
    *reinterpret_cast<v8bf*>(dst)     = *reinterpret_cast<v8bf*>(tmp);
    *reinterpret_cast<v8bf*>(dst + 8) = *reinterpret_cast<v8bf*>(tmp + 8);
}

// x (B,81,IN) f32 -> bf16 xb[(b*81+n)][i], i padded to 64
__global__ void convert_x_kernel(const float* __restrict__ x, bf16_t* __restrict__ xb, int in_dim) {
    size_t tid = (size_t)blockIdx.x * blockDim.x + threadIdx.x;   // B*81*64
    int i = (int)(tid & 63);
    size_t row = tid >> 6;
    float v = (i < in_dim) ? x[row * in_dim + i] : 0.0f;
    xb[row * 64 + i] = (bf16_t)v;
}

__global__ void zero_stats_kernel(float* p) {
    if (threadIdx.x < 2 * NPAD) p[threadIdx.x] = 0.0f;
}

// ---------------------------------------------------------------------------
// Kernel A: per-node GEMM  T[b,n,o] = x[b,n,:] @ W[n]
// 8 waves/WG, one node per wave; wave tile = 16 rows(b) x 64 cols(o) with
// 4 accumulators (A-frag reuse x4). Epilogue repacks through a 16KB LDS tile
// [16m][64o][8n] so global stores are contiguous 16B chunks spanning 8 nodes
// of the transposed tt[b][o][n] layout (B-matrix of the mix GEMM).
// ---------------------------------------------------------------------------
template <int K>
__global__ __launch_bounds__(256) void node_gemm_kernel(
    const bf16_t* __restrict__ xb,   // [B][81][K]
    const bf16_t* __restrict__ wt,   // [81][256][K]
    bf16_t* __restrict__ tt)         // [B][256][96]
{
    __shared__ bf16_t lds[16 * 64 * 8];          // [m][o][n] bf16, 16KB

    const int lane = threadIdx.x & 31;
    const int wv   = threadIdx.x >> 5;
    const int o0   = blockIdx.x * 64;
    const int b0   = blockIdx.y * 16;
    const int nb   = blockIdx.z * 8;             // node base for this WG
    const int n    = nb + wv;                    // this wave's node

    const int lo16 = lane & 15;
    const int hi   = lane >> 4;

    v8f acc[4] = {v8f{}, v8f{}, v8f{}, v8f{}};

    if (n < NNODE) {
        const bf16_t* xrow  = xb + ((size_t)(b0 + lo16) * NNODE + n) * K;  // A: lane = M row
        const bf16_t* wbase = wt + (size_t)n * EMB * K;
        #pragma unroll
        for (int kc = 0; kc < K; kc += 32) {
            const int ka = kc + hi * 16;         // A-frag: 16 contiguous K per lane
            v16bf afrag = load_frag16(xrow + ka, xrow + ka + 8);
            const int kb = kc + hi * 16;         // B-frag: 16 contiguous K per lane
            #pragma unroll
            for (int t = 0; t < 4; ++t) {
                const bf16_t* wp = wbase + (size_t)(o0 + t * 16 + lo16) * K + kb;
                v16bf bfrag = load_frag16(wp, wp + 8);
                acc[t] = __builtin_amdgcn_wmma_f32_16x16x32_bf16(
                    false, afrag, false, bfrag, (short)0, acc[t], false, false);
            }
        }
    }
    // (invalid waves keep acc == 0 -> deterministic zero padding in tt)

    #pragma unroll
    for (int t = 0; t < 4; ++t) {
        const int ol = t * 16 + lo16;
        #pragma unroll
        for (int r = 0; r < 8; ++r) {
            const int m = hi * 8 + r;            // C/D layout: VGPR r -> M = 8*hi + r
            lds[(m * 64 + ol) * 8 + wv] = (bf16_t)acc[t][r];
        }
    }
    __syncthreads();

    // 1024 (m,o) elements; each gets 8 consecutive n -> one 16B store.
    for (int e = threadIdx.x; e < 16 * 64; e += 256) {
        const int m  = e >> 6;
        const int ol = e & 63;
        v8bf v = *reinterpret_cast<v8bf*>(&lds[e * 8]);
        *reinterpret_cast<v8bf*>(
            &tt[((size_t)(b0 + m) * EMB + o0 + ol) * NPAD + nb]) = v;
    }
}

// ---------------------------------------------------------------------------
// Kernel B: adjacency mix  y[b,m,o] = sum_n Adj[m,n]*T[b,n,o] + bias[o]
// as bf16 WMMA GEMM (M=96 pad, K=96 pad, N=256). One wave per (b, o-tile)
// holds all 6 m-tiles (6 accumulators) so each tt B-fragment is loaded ONCE;
// adjacency A-fragments are 18KB total and stay cache-resident.
// Fused BN partial stats via shfl-xor + global f32 atomics.
// ---------------------------------------------------------------------------
__global__ __launch_bounds__(256) void mix_stats_kernel(
    const bf16_t* __restrict__ adjb,  // [96][96]
    const bf16_t* __restrict__ tt,    // [B][256][96]
    const float*  __restrict__ bias,  // [256]
    float* __restrict__ y,            // [B][81][256]
    float* __restrict__ ssum,         // [96]
    float* __restrict__ ssq)          // [96]
{
    const int lane = threadIdx.x & 31;
    const int id   = blockIdx.x * 8 + (threadIdx.x >> 5);   // (b, ot)
    const int b    = id >> 4;
    const int ot   = id & 15;
    const int lo16 = lane & 15;
    const int hi   = lane >> 4;

    v8f acc[6] = {v8f{}, v8f{}, v8f{}, v8f{}, v8f{}, v8f{}};

    const bf16_t* trow = tt + ((size_t)b * EMB + ot * 16 + lo16) * NPAD;
    #pragma unroll
    for (int kc = 0; kc < NPAD; kc += 32) {
        const int kb = kc + hi * 16;
        v16bf bfrag = load_frag16(trow + kb, trow + kb + 8);   // loaded once
        const int ka = kc + hi * 16;
        #pragma unroll
        for (int j = 0; j < 6; ++j) {
            const bf16_t* arow = adjb + (size_t)(j * 16 + lo16) * NPAD;
            v16bf afrag = load_frag16(arow + ka, arow + ka + 8);
            acc[j] = __builtin_amdgcn_wmma_f32_16x16x32_bf16(
                false, afrag, false, bfrag, (short)0, acc[j], false, false);
        }
    }

    const int oc = ot * 16 + lo16;
    const float bv = bias[oc];
    #pragma unroll
    for (int j = 0; j < 6; ++j) {
        float s[8], q[8];
        #pragma unroll
        for (int r = 0; r < 8; ++r) {
            const int m = j * 16 + hi * 8 + r;
            const float v = acc[j][r] + bv;
            if (m < NNODE) {
                y[((size_t)b * NNODE + m) * EMB + oc] = v;
                s[r] = v; q[r] = v * v;
            } else { s[r] = 0.0f; q[r] = 0.0f; }
        }
        // reduce across the 16 lanes sharing each row set (xor masks < 16)
        #pragma unroll
        for (int off = 8; off > 0; off >>= 1) {
            #pragma unroll
            for (int r = 0; r < 8; ++r) {
                s[r] += __shfl_xor(s[r], off, 32);
                q[r] += __shfl_xor(q[r], off, 32);
            }
        }
        if (lo16 == 0) {
            #pragma unroll
            for (int r = 0; r < 8; ++r) {
                const int m = j * 16 + hi * 8 + r;
                if (m < NNODE) {
                    atomicAdd(&ssum[m], s[r]);
                    atomicAdd(&ssq[m],  q[r]);
                }
            }
        }
    }
}

// ---------------------------------------------------------------------------
// Kernel C: finalize BN stats -> per-node scale/shift
// ---------------------------------------------------------------------------
__global__ void finalize_stats_kernel(
    const float* __restrict__ ssum, const float* __restrict__ ssq,
    const float* __restrict__ gamma, const float* __restrict__ beta,
    float* __restrict__ scale, float* __restrict__ shift)
{
    int m = threadIdx.x;
    if (m < NNODE) {
        const float inv_cnt = 1.0f / ((float)BATCH * (float)EMB);
        float mu  = ssum[m] * inv_cnt;
        float var = ssq[m] * inv_cnt - mu * mu;
        float rs  = rsqrtf(var + 1e-5f);
        float sc  = gamma[m] * rs;
        scale[m] = sc;
        shift[m] = beta[m] - mu * sc;
    }
}

// ---------------------------------------------------------------------------
// Kernel D: BN affine + ReLU + residual; in-place f32 update + bf16 copy for
// next GEMM. Last block streams into d_out (stride 512, half offset applied).
// ---------------------------------------------------------------------------
__global__ __launch_bounds__(256) void bn_relu_res_kernel(
    const float* __restrict__ y,
    const float* __restrict__ scale, const float* __restrict__ shift,
    float* __restrict__ xio,          // [B][81][256] residual in / f32 out
    bf16_t* __restrict__ xb,          // [B][81][256] bf16 out (unused if last)
    float* __restrict__ outp,         // d_out + half*256, or nullptr
    int res)
{
    size_t tid = (size_t)blockIdx.x * blockDim.x + threadIdx.x;  // B*81*64
    const int o4 = (int)(tid & 63) * 4;
    const size_t bm = tid >> 6;                                  // b*81+m
    const int m = (int)(bm % NNODE);

    const float4 yv = *reinterpret_cast<const float4*>(y + bm * EMB + o4);
    const float sc = scale[m], sh = shift[m];
    float4 r;
    r.x = fmaxf(yv.x * sc + sh, 0.0f);
    r.y = fmaxf(yv.y * sc + sh, 0.0f);
    r.z = fmaxf(yv.z * sc + sh, 0.0f);
    r.w = fmaxf(yv.w * sc + sh, 0.0f);
    if (res) {
        const float4 xv = *reinterpret_cast<const float4*>(xio + bm * EMB + o4);
        r.x += xv.x; r.y += xv.y; r.z += xv.z; r.w += xv.w;
    }
    if (outp) {
        *reinterpret_cast<float4*>(outp + bm * 512 + o4) = r;
    } else {
        *reinterpret_cast<float4*>(xio + bm * EMB + o4) = r;
        bf16_t* xp = xb + bm * EMB + o4;
        xp[0] = (bf16_t)r.x; xp[1] = (bf16_t)r.y;
        xp[2] = (bf16_t)r.z; xp[3] = (bf16_t)r.w;
    }
}

// ---------------------------------------------------------------------------
// Host-side orchestration
// ---------------------------------------------------------------------------
extern "C" void kernel_launch(void* const* d_in, const int* in_sizes, int n_in,
                              void* d_out, int out_size, void* d_ws, size_t ws_size,
                              hipStream_t stream) {
    (void)in_sizes; (void)n_in; (void)out_size; (void)ws_size;

    // workspace partition (all offsets 256B-aligned)
    char* ws = (char*)d_ws;
    const size_t ADJ_B  = (size_t)NPAD * NPAD * 2;                    //     18,432
    const size_t W0T_B  = (size_t)NNODE * EMB * 64 * 2;               //  2,654,208
    const size_t WT_B   = (size_t)7 * NNODE * EMB * EMB * 2;          // 74,317,824
    const size_t XB_B   = (size_t)BATCH * NNODE * EMB * 2;            // 84,934,656
    const size_t TT_B   = (size_t)BATCH * EMB * NPAD * 2;             //100,663,296
    const size_t Y_B    = (size_t)BATCH * NNODE * EMB * 4;            //169,869,312

    bf16_t* adjb = (bf16_t*)ws;                      ws += ADJ_B;
    bf16_t* w0t  = (bf16_t*)ws;                      ws += W0T_B;
    bf16_t* wt   = (bf16_t*)ws;                      ws += WT_B;
    bf16_t* xbuf = (bf16_t*)ws;                      ws += XB_B;
    bf16_t* ttb  = (bf16_t*)ws;                      ws += TT_B;
    float*  ybuf = (float*)ws;                       ws += Y_B;
    float*  xf   = (float*)ws;                       ws += Y_B;
    float*  ssum = (float*)ws;
    float*  ssq  = ssum + NPAD;
    float*  scl  = ssum + 2 * NPAD;
    float*  shf  = ssum + 3 * NPAD;

    const float* Aadj = (const float*)d_in[2];
    convert_adj_kernel<<<(NPAD * NPAD + 255) / 256, 256, 0, stream>>>(Aadj, adjb);

    const int BMROWS = BATCH * NNODE;                 // 165,888
    const int GRID64 = BMROWS * 64 / 256;             // 41,472
    const int GRIDMX = BATCH * 16 / 8;                //  4,096

    for (int h = 0; h < 2; ++h) {
        const float* x     = (const float*)d_in[h == 0 ? 0 : 1];
        const float* W0    = (const float*)d_in[h == 0 ? 3 : 8];
        const float* W     = (const float*)d_in[h == 0 ? 4 : 9];
        const float* bias  = (const float*)d_in[h == 0 ? 5 : 10];
        const float* gamma = (const float*)d_in[h == 0 ? 6 : 11];
        const float* beta  = (const float*)d_in[h == 0 ? 7 : 12];
        const int in_dim   = (h == 0) ? 35 : 40;

        convert_w0_kernel<<<NNODE, 256, 0, stream>>>(W0, w0t, in_dim);
        convert_w_kernel<<<dim3(16, 7 * NNODE), 256, 0, stream>>>(W, wt);
        convert_x_kernel<<<GRID64, 256, 0, stream>>>(x, xbuf, in_dim);

        for (int k = 0; k < NBLK; ++k) {
            zero_stats_kernel<<<1, 256, 0, stream>>>(ssum);

            if (k == 0) {
                node_gemm_kernel<64><<<dim3(4, 128, 11), 256, 0, stream>>>(xbuf, w0t, ttb);
            } else {
                const bf16_t* wk = wt + (size_t)(k - 1) * NNODE * EMB * EMB;
                node_gemm_kernel<256><<<dim3(4, 128, 11), 256, 0, stream>>>(xbuf, wk, ttb);
            }

            mix_stats_kernel<<<GRIDMX, 256, 0, stream>>>(
                adjb, ttb, bias + k * EMB, ybuf, ssum, ssq);

            finalize_stats_kernel<<<1, NPAD, 0, stream>>>(
                ssum, ssq, gamma + k * NNODE, beta + k * NNODE, scl, shf);

            float* outp = (k == NBLK - 1) ? ((float*)d_out + h * EMB) : nullptr;
            bn_relu_res_kernel<<<GRID64, 256, 0, stream>>>(
                ybuf, scl, shf, xf, xbuf, outp, (k > 0) ? 1 : 0);
        }
    }
}
